// SpaceNonLocal_8538394984581
// MI455X (gfx1250) — matrix operand order repaired and checked by reading, not compiled, not run on hardware
//
#include <hip/hip_runtime.h>

// ---------------------------------------------------------------------------
// CDNA5 (gfx1250) bf16 WMMA implementation of the non-local attention block.
// Shapes: B=8, C=1024, H=W=64 (N=4096), Ci=512, pooled M=1024.
// 64x32 register-blocked GEMMs, 2x K-step double buffering, 32-bit lane
// offsets (saddr-form global loads), LDS-resident softmax.
// ---------------------------------------------------------------------------

typedef __attribute__((ext_vector_type(16))) __bf16 v16bf;
typedef __attribute__((ext_vector_type(8)))  float  v8f;

union FragB { v16bf v; uint4 q[2]; };
union Pack8 { uint4 q; unsigned short h[8]; };

__device__ __forceinline__ unsigned short f2bf(float f) {
  unsigned int u = __float_as_uint(f);
  u += 0x7FFFu + ((u >> 16) & 1u);           // round-to-nearest-even
  return (unsigned short)(u >> 16);
}
__device__ __forceinline__ float bf2f(unsigned short h) {
  return __uint_as_float(((unsigned int)h) << 16);
}

// A-matrix 16x32 bf16 fragment (row-major source [rows, ld]); 32-bit offsets.
// ISA layout: lane<16 row=lane, K={k0..k0+7, k0+16..k0+23};
//             lane>=16 same row, K shifted by +8.
__device__ __forceinline__ v16bf load_a_frag(const unsigned short* base,
                                             int row0, int k0, int ld, int lane) {
  int r   = row0 + (lane & 15);
  int kk  = k0 + ((lane >> 4) << 3);
  int idx = r * ld + kk;                     // per-batch slices < 2^31 elements
  FragB f;
  f.q[0] = *(const uint4*)(base + idx);
  f.q[1] = *(const uint4*)(base + idx + 16);
  return f.v;
}

// B-matrix 32x16 bf16 fragment from a column-contiguous ("T") source:
// element (k, n) lives at base[n*ld + k]; 32-bit offsets.
// ISA layout: lane holds column n = n0 + (lane&15); lane<16 K={k0..k0+7, k0+16..k0+23},
//             lane>=16 K shifted by +8 (matches load_a_frag).
__device__ __forceinline__ v16bf load_bT_frag(const unsigned short* base,
                                              int k0, int n0, int ld, int lane) {
  int n   = n0 + (lane & 15);
  int kk  = k0 + ((lane >> 4) << 3);
  int idx = n * ld + kk;
  FragB f;
  f.q[0] = *(const uint4*)(base + idx);
  f.q[1] = *(const uint4*)(base + idx + 16);
  return f.v;
}

__device__ __forceinline__ v8f wmma_bf16(v16bf a, v16bf b, v8f c) {
  return __builtin_amdgcn_wmma_f32_16x16x32_bf16(false, a, false, b,
                                                 (short)0, c, false, false);
}

// ---------------------------------------------------------------------------
// K1: fp32 weights -> bf16
// ---------------------------------------------------------------------------
__global__ void cvt_weight_kernel(const float* __restrict__ w,
                                  unsigned short* __restrict__ out, int n) {
  int i = blockIdx.x * blockDim.x + threadIdx.x;
  if (i < n) out[i] = f2bf(w[i]);
}

// ---------------------------------------------------------------------------
// K2a: x[B,C,N] fp32 -> xT[B,N,C] bf16 (LDS-tiled transpose, coalesced both sides)
// ---------------------------------------------------------------------------
__global__ __launch_bounds__(256)
void transpose_x_kernel(const float* __restrict__ x, unsigned short* __restrict__ xT) {
  __shared__ unsigned short tile[32][33];
  const int C = 1024, N = 4096;
  int b = blockIdx.z;
  int cblk = blockIdx.y * 32;
  int nblk = blockIdx.x * 32;
  int tx = threadIdx.x, ty = threadIdx.y;           // block (32, 8)
#pragma unroll
  for (int i = 0; i < 4; ++i) {
    int c = cblk + ty + i * 8;
    tile[ty + i * 8][tx] = f2bf(x[((size_t)b * C + c) * N + nblk + tx]);
  }
  __syncthreads();
#pragma unroll
  for (int i = 0; i < 4; ++i) {
    int n = nblk + ty + i * 8;
    xT[((size_t)b * N + n) * C + cblk + tx] = tile[tx][ty + i * 8];
  }
}

// ---------------------------------------------------------------------------
// K2b: 2x2 maxpool from xT -> pxT[B,M,C] bf16 (coalesced in C)
// ---------------------------------------------------------------------------
__global__ void pool_kernel(const unsigned short* __restrict__ xT,
                            unsigned short* __restrict__ pxT) {
  const int C = 1024, N = 4096, M = 1024;
  size_t idx = (size_t)blockIdx.x * blockDim.x + threadIdx.x;   // B*M*C threads
  int c = idx & (C - 1);
  int m = (idx >> 10) & (M - 1);
  int b = idx >> 20;
  int h2 = m >> 5, w2 = m & 31;
  int n00 = (h2 * 2) * 64 + w2 * 2;
  const unsigned short* p = xT + ((size_t)b * N) * C + c;
  float v0 = bf2f(p[(n00)      * C]);
  float v1 = bf2f(p[(n00 + 1)  * C]);
  float v2 = bf2f(p[(n00 + 64) * C]);
  float v3 = bf2f(p[(n00 + 65) * C]);
  float mx = fmaxf(fmaxf(v0, v1), fmaxf(v2, v3));
  pxT[((size_t)b * M + m) * C + c] = f2bf(mx);
}

// ---------------------------------------------------------------------------
// K3: bf16 GEMM, 64(O) x 32(N) per wave; two K-steps in flight per iteration.
// D[o,n] = scale * (sum_k W[o,k]*XT[n,k] + bias[o]).
// store_transposed: out[B,N,O] bf16, else out[B,O,N] bf16.
// ---------------------------------------------------------------------------
__global__ __launch_bounds__(128)
void gemm_bf16_kernel(const unsigned short* __restrict__ W,
                      const float* __restrict__ bias,
                      const unsigned short* __restrict__ XT,
                      unsigned short* __restrict__ out,
                      int O, int N, int K, float scale, int store_transposed) {
  int lane = threadIdx.x & 31;
  int wave = threadIdx.x >> 5;
  int n0 = (blockIdx.x * 4 + wave) * 32;
  int o0 = blockIdx.y * 64;
  int b  = blockIdx.z;
  const unsigned short* xt = XT + (size_t)b * N * K;

  v8f acc[4][2];
#pragma unroll
  for (int ot = 0; ot < 4; ++ot)
#pragma unroll
    for (int nt = 0; nt < 2; ++nt)
      acc[ot][nt] = (v8f){0.f, 0.f, 0.f, 0.f, 0.f, 0.f, 0.f, 0.f};

  for (int k0 = 0; k0 < K; k0 += 64) {
    v16bf a0[4], b0[2], a1[4], b1[2];
#pragma unroll
    for (int ot = 0; ot < 4; ++ot) a0[ot] = load_a_frag(W, o0 + ot * 16, k0, K, lane);
#pragma unroll
    for (int nt = 0; nt < 2; ++nt) b0[nt] = load_bT_frag(xt, k0, n0 + nt * 16, K, lane);
#pragma unroll
    for (int ot = 0; ot < 4; ++ot) a1[ot] = load_a_frag(W, o0 + ot * 16, k0 + 32, K, lane);
#pragma unroll
    for (int nt = 0; nt < 2; ++nt) b1[nt] = load_bT_frag(xt, k0 + 32, n0 + nt * 16, K, lane);
#pragma unroll
    for (int ot = 0; ot < 4; ++ot)
#pragma unroll
      for (int nt = 0; nt < 2; ++nt)
        acc[ot][nt] = wmma_bf16(a0[ot], b0[nt], acc[ot][nt]);
#pragma unroll
    for (int ot = 0; ot < 4; ++ot)
#pragma unroll
      for (int nt = 0; nt < 2; ++nt)
        acc[ot][nt] = wmma_bf16(a1[ot], b1[nt], acc[ot][nt]);
  }

  int hi = (lane >> 4) << 3;           // 0 or 8
#pragma unroll
  for (int nt = 0; nt < 2; ++nt) {
    int n = n0 + nt * 16 + (lane & 15);
    if (store_transposed) {
#pragma unroll
      for (int ot = 0; ot < 4; ++ot) {
        Pack8 pk;
#pragma unroll
        for (int v = 0; v < 8; ++v) {
          int o = o0 + ot * 16 + v + hi;
          pk.h[v] = f2bf((acc[ot][nt][v] + bias[o]) * scale);
        }
        *(uint4*)&out[((size_t)b * N + n) * O + o0 + ot * 16 + hi] = pk.q;
      }
    } else {
#pragma unroll
      for (int ot = 0; ot < 4; ++ot)
#pragma unroll
        for (int v = 0; v < 8; ++v) {
          int o = o0 + ot * 16 + v + hi;
          out[((size_t)b * O + o) * N + n] = f2bf((acc[ot][nt][v] + bias[o]) * scale);
        }
    }
  }
}

// ---------------------------------------------------------------------------
// K4: fused attention. One wave per 16-query tile.
//   scores(16x1024) -> LDS fp32 -> softmax -> probs bf16 in LDS ->
//   t = g @ P^T with 4 c-tiles sharing each LDS P-fragment -> tT[B,N,Ci] bf16.
// ---------------------------------------------------------------------------
#define ATT_SLD 1026   // padded fp32 score row stride (bank-conflict free)
#define ATT_PLD 1032   // padded bf16 prob row stride (16B aligned, 4-bank skew)
#define ATT_SMEM (16 * ATT_SLD * 4 + 16 * ATT_PLD * 2 + 64)

__global__ __launch_bounds__(32)
void attention_kernel(const unsigned short* __restrict__ thetaT, // [B,N,Ci]
                      const unsigned short* __restrict__ phiT,   // [B,M,Ci]
                      const unsigned short* __restrict__ G,      // [B,Ci,M]
                      unsigned short* __restrict__ tT,           // [B,N,Ci]
                      int N, int M, int Ci) {
  extern __shared__ char smem_raw[];
  float*          S  = (float*)smem_raw;                               // 16 x ATT_SLD
  unsigned short* P  = (unsigned short*)(smem_raw + 16 * ATT_SLD * 4); // 16 x ATT_PLD
  float*          RS = (float*)(smem_raw + 16 * ATT_SLD * 4 + 16 * ATT_PLD * 2);

  int lane = threadIdx.x & 31;
  int n0 = blockIdx.x * 16;
  int b  = blockIdx.y;
  const unsigned short* th = thetaT + (size_t)b * N * Ci;
  const unsigned short* ph = phiT   + (size_t)b * M * Ci;
  const unsigned short* g  = G      + (size_t)b * Ci * M;
  unsigned short*       to = tT     + (size_t)b * N * Ci;

  // Preload all Q A-fragments for Ci=512 (16 K-steps, 128 VGPRs) - reused 64x.
  v16bf qf[16];
#pragma unroll
  for (int kc = 0; kc < 16; ++kc) qf[kc] = load_a_frag(th, n0, kc * 32, Ci, lane);

  int hi = (lane >> 4) << 3;

  // Phase 1: scores = Q K^T into LDS; two key chunks in flight per iteration.
  for (int mc = 0; mc < M / 32; ++mc) {
    int m0 = mc * 32;
    v8f acc0 = {0.f, 0.f, 0.f, 0.f, 0.f, 0.f, 0.f, 0.f};
    v8f acc1 = {0.f, 0.f, 0.f, 0.f, 0.f, 0.f, 0.f, 0.f};
#pragma unroll
    for (int kc = 0; kc < 16; ++kc) {
      v16bf bb0 = load_bT_frag(ph, kc * 32, m0,      Ci, lane);
      v16bf bb1 = load_bT_frag(ph, kc * 32, m0 + 16, Ci, lane);
      acc0 = wmma_bf16(qf[kc], bb0, acc0);
      acc1 = wmma_bf16(qf[kc], bb1, acc1);
    }
#pragma unroll
    for (int v = 0; v < 8; ++v) {
      S[(v + hi) * ATT_SLD + m0      + (lane & 15)] = acc0[v];
      S[(v + hi) * ATT_SLD + m0 + 16 + (lane & 15)] = acc1[v];
    }
  }
  __syncthreads();

  // Softmax over M per row: 2 lanes per row.
  {
    int row = lane >> 1, half = lane & 1;
    float mx = -3.4e38f;
    for (int m = half; m < M; m += 2) mx = fmaxf(mx, S[row * ATT_SLD + m]);
    mx = fmaxf(mx, __shfl_xor(mx, 1));
    float sum = 0.f;
    for (int m = half; m < M; m += 2) {
      float e = __expf(S[row * ATT_SLD + m] - mx);
      P[row * ATT_PLD + m] = f2bf(e);
      sum += e;
    }
    sum += __shfl_xor(sum, 1);
    if (half == 0) RS[row] = 1.0f / sum;
  }
  __syncthreads();

  // Phase 2: t[c, n] = sum_m g[c, m] * P[n, m]; 4 c-tiles share each P-frag.
  float rs = RS[lane & 15];
  for (int cg = 0; cg < Ci / 64; ++cg) {        // 8 groups of 4 c-tiles
    v8f acc[4];
#pragma unroll
    for (int j = 0; j < 4; ++j)
      acc[j] = (v8f){0.f, 0.f, 0.f, 0.f, 0.f, 0.f, 0.f, 0.f};
    for (int mc = 0; mc < M / 32; ++mc) {
      int m0 = mc * 32;
      v16bf bb = load_bT_frag(P, m0, 0, ATT_PLD, lane);  // ds_load_b128 x2
#pragma unroll
      for (int j = 0; j < 4; ++j) {
        v16bf a = load_a_frag(g, cg * 64 + j * 16, m0, M, lane);
        acc[j] = wmma_bf16(a, bb, acc[j]);
      }
    }
    int n = n0 + (lane & 15);
#pragma unroll
    for (int j = 0; j < 4; ++j) {
      int c0 = cg * 64 + j * 16;
      Pack8 pk;
#pragma unroll
      for (int v = 0; v < 8; ++v) pk.h[v] = f2bf(acc[j][v] * rs);
      *(uint4*)&to[(size_t)n * Ci + c0 + hi] = pk.q;
    }
  }
}

// ---------------------------------------------------------------------------
// K5: output conv: Y[b,o,n] = sum_k Wout[o,k] * tT[b,n,k] + bias[o]  (fp32 out)
// 64(O) x 32(N) per wave; two K-steps in flight per iteration.
// ---------------------------------------------------------------------------
__global__ __launch_bounds__(128)
void gemm_out_kernel(const unsigned short* __restrict__ W,
                     const float* __restrict__ bias,
                     const unsigned short* __restrict__ TT,
                     float* __restrict__ Y, int O, int N, int K) {
  int lane = threadIdx.x & 31;
  int wave = threadIdx.x >> 5;
  int n0 = (blockIdx.x * 4 + wave) * 32;
  int o0 = blockIdx.y * 64;
  int b  = blockIdx.z;
  const unsigned short* tt = TT + (size_t)b * N * K;

  v8f acc[4][2];
#pragma unroll
  for (int ot = 0; ot < 4; ++ot)
#pragma unroll
    for (int nt = 0; nt < 2; ++nt)
      acc[ot][nt] = (v8f){0.f, 0.f, 0.f, 0.f, 0.f, 0.f, 0.f, 0.f};

  for (int k0 = 0; k0 < K; k0 += 64) {
    v16bf a0[4], b0[2], a1[4], b1[2];
#pragma unroll
    for (int ot = 0; ot < 4; ++ot) a0[ot] = load_a_frag(W, o0 + ot * 16, k0, K, lane);
#pragma unroll
    for (int nt = 0; nt < 2; ++nt) b0[nt] = load_bT_frag(tt, k0, n0 + nt * 16, K, lane);
#pragma unroll
    for (int ot = 0; ot < 4; ++ot) a1[ot] = load_a_frag(W, o0 + ot * 16, k0 + 32, K, lane);
#pragma unroll
    for (int nt = 0; nt < 2; ++nt) b1[nt] = load_bT_frag(tt, k0 + 32, n0 + nt * 16, K, lane);
#pragma unroll
    for (int ot = 0; ot < 4; ++ot)
#pragma unroll
      for (int nt = 0; nt < 2; ++nt)
        acc[ot][nt] = wmma_bf16(a0[ot], b0[nt], acc[ot][nt]);
#pragma unroll
    for (int ot = 0; ot < 4; ++ot)
#pragma unroll
      for (int nt = 0; nt < 2; ++nt)
        acc[ot][nt] = wmma_bf16(a1[ot], b1[nt], acc[ot][nt]);
  }

  int hi = (lane >> 4) << 3;
#pragma unroll
  for (int nt = 0; nt < 2; ++nt) {
    int n = n0 + nt * 16 + (lane & 15);
#pragma unroll
    for (int ot = 0; ot < 4; ++ot)
#pragma unroll
      for (int v = 0; v < 8; ++v) {
        int o = o0 + ot * 16 + v + hi;
        Y[((size_t)b * O + o) * N + n] = acc[ot][nt][v] + bias[o];
      }
  }
}

// ---------------------------------------------------------------------------
// K6: per-channel BN stats (one block per channel, no atomics)
// ---------------------------------------------------------------------------
__global__ __launch_bounds__(256)
void stats_kernel(const float* __restrict__ Y, float* __restrict__ stats) {
  const int C = 1024, N = 4096, B = 8;
  __shared__ float s1[256], s2[256];
  int c = blockIdx.x;
  float sum = 0.f, sq = 0.f;
  for (int i = threadIdx.x; i < B * N; i += 256) {
    int bb = i >> 12, n = i & (N - 1);
    float v = Y[((size_t)bb * C + c) * N + n];
    sum += v; sq += v * v;
  }
  s1[threadIdx.x] = sum; s2[threadIdx.x] = sq;
  __syncthreads();
  for (int s = 128; s > 0; s >>= 1) {
    if (threadIdx.x < s) { s1[threadIdx.x] += s1[threadIdx.x + s];
                           s2[threadIdx.x] += s2[threadIdx.x + s]; }
    __syncthreads();
  }
  if (threadIdx.x == 0) { stats[c] = s1[0]; stats[C + c] = s2[0]; }
}

// ---------------------------------------------------------------------------
// K7: finalize: BN(train) + gamma/beta + residual
// ---------------------------------------------------------------------------
__global__ void finalize_kernel(const float* __restrict__ Y,
                                const float* __restrict__ x,
                                const float* __restrict__ stats,
                                const float* __restrict__ gamma,
                                const float* __restrict__ beta,
                                float* __restrict__ out) {
  const int C = 1024;
  const float inv_cnt = 1.0f / 32768.0f;  // B*N
  size_t idx = (size_t)blockIdx.x * blockDim.x + threadIdx.x;
  int c = (idx >> 12) & (C - 1);
  float mean = stats[c] * inv_cnt;
  float var  = stats[C + c] * inv_cnt - mean * mean;
  float y = (Y[idx] - mean) * rsqrtf(var + 1e-5f) * gamma[c] + beta[c];
  out[idx] = y + x[idx];
}

// ---------------------------------------------------------------------------
// Host launcher
// ---------------------------------------------------------------------------
extern "C" void kernel_launch(void* const* d_in, const int* in_sizes, int n_in,
                              void* d_out, int out_size, void* d_ws, size_t ws_size,
                              hipStream_t stream) {
  (void)in_sizes; (void)n_in; (void)out_size; (void)ws_size;
  const int B = 8, C = 1024, N = 4096, Ci = 512, M = 1024;

  const float* x       = (const float*)d_in[0];
  const float* w_theta = (const float*)d_in[1];
  const float* b_theta = (const float*)d_in[2];
  const float* w_phi   = (const float*)d_in[3];
  const float* b_phi   = (const float*)d_in[4];
  const float* w_g     = (const float*)d_in[5];
  const float* b_g     = (const float*)d_in[6];
  const float* w_out   = (const float*)d_in[7];
  const float* b_out   = (const float*)d_in[8];
  const float* gamma   = (const float*)d_in[9];
  const float* beta    = (const float*)d_in[10];

  char* ws = (char*)d_ws;
  size_t off = 0;
  auto alloc = [&](size_t bytes) { char* p = ws + off; off = (off + bytes + 255) & ~(size_t)255; return p; };
  unsigned short* xT     = (unsigned short*)alloc((size_t)B * N * C * 2);   //  64 MB
  unsigned short* pxT    = (unsigned short*)alloc((size_t)B * M * C * 2);   //  16 MB
  unsigned short* wth    = (unsigned short*)alloc((size_t)Ci * C * 2);
  unsigned short* wph    = (unsigned short*)alloc((size_t)Ci * C * 2);
  unsigned short* wg     = (unsigned short*)alloc((size_t)Ci * C * 2);
  unsigned short* wout   = (unsigned short*)alloc((size_t)C * Ci * 2);
  unsigned short* thetaT = (unsigned short*)alloc((size_t)B * N * Ci * 2);  //  32 MB
  unsigned short* phiT   = (unsigned short*)alloc((size_t)B * M * Ci * 2);  //   8 MB
  unsigned short* gbuf   = (unsigned short*)alloc((size_t)B * Ci * M * 2);  //   8 MB
  unsigned short* tT     = (unsigned short*)alloc((size_t)B * N * Ci * 2);  //  32 MB
  float*          Y      = (float*)alloc((size_t)B * C * N * 4);            // 128 MB
  float*          stats  = (float*)alloc((size_t)2 * C * 4);

  // K1: weights -> bf16
  cvt_weight_kernel<<<(Ci * C + 255) / 256, 256, 0, stream>>>(w_theta, wth, Ci * C);
  cvt_weight_kernel<<<(Ci * C + 255) / 256, 256, 0, stream>>>(w_phi,   wph, Ci * C);
  cvt_weight_kernel<<<(Ci * C + 255) / 256, 256, 0, stream>>>(w_g,     wg,  Ci * C);
  cvt_weight_kernel<<<(C * Ci + 255) / 256, 256, 0, stream>>>(w_out,   wout, C * Ci);

  // K2: transpose + pool
  transpose_x_kernel<<<dim3(N / 32, C / 32, B), dim3(32, 8), 0, stream>>>(x, xT);
  pool_kernel<<<(size_t)B * M * C / 256, 256, 0, stream>>>(xT, pxT);

  // K3: theta (scaled by Ci^-1/2), phi, g — 64x32 tiles per wave
  const float inv_sqrt_ci = 0.04419417382415922f;
  gemm_bf16_kernel<<<dim3(N / 128, Ci / 64, B), 128, 0, stream>>>(
      wth, b_theta, xT, thetaT, Ci, N, C, inv_sqrt_ci, 1);
  gemm_bf16_kernel<<<dim3(M / 128, Ci / 64, B), 128, 0, stream>>>(
      wph, b_phi, pxT, phiT, Ci, M, C, 1.0f, 1);
  gemm_bf16_kernel<<<dim3(M / 128, Ci / 64, B), 128, 0, stream>>>(
      wg, b_g, pxT, gbuf, Ci, M, C, 1.0f, 0);

  // K4: fused attention (softmax resident in LDS)
  attention_kernel<<<dim3(N / 16, B), 32, ATT_SMEM, stream>>>(
      thetaT, phiT, gbuf, tT, N, M, Ci);

  // K5: output projection — 64x32 tiles per wave
  gemm_out_kernel<<<dim3(N / 128, C / 64, B), 128, 0, stream>>>(
      wout, b_out, tT, Y, C, N, Ci);

  // K6 + K7: batch-norm stats, then normalize + residual
  stats_kernel<<<C, 256, 0, stream>>>(Y, stats);
  finalize_kernel<<<(size_t)B * C * N / 256, 256, 0, stream>>>(
      Y, x, stats, gamma, beta, (float*)d_out);
}
